// MaskBasedRNNLayer_21698174779457
// MI455X (gfx1250) — matrix operand order repaired and checked by reading, not compiled, not run on hardware
//
#include <hip/hip_runtime.h>

// ---------------------------------------------------------------------------
// Bidirectional masked LSTM for MI455X (gfx1250, wave32, WMMA).
//   B=64, T=512, D=256, HC=128, G=4*HC=512
// Phase 1: convert x / Wih / Whh to bf16 (workspace).
// Phase 2: xg[dir] = x @ Wih^T + bih + bhh   (v_wmma_f32_16x16x32_bf16)
// Phase 3: sequential scan, 8 workgroups (2 dirs x 4 batch-slices of 16),
//          Whh B-fragments pinned in VGPRs, h state in LDS (bf16),
//          c/h f32 state in registers, WMMA per step.
// ---------------------------------------------------------------------------

#define LSTM_B   64
#define LSTM_T   512
#define LSTM_D   256
#define LSTM_HC  128
#define LSTM_G   512          // 4*HC
#define GSTR     516          // gates LDS row stride (f32), 516%64=4 -> conflict-free
#define HSTR     134          // h LDS row stride (bf16), dword stride 67 -> conflict-free

typedef __attribute__((ext_vector_type(16))) __bf16 v16bf;
typedef __attribute__((ext_vector_type(8)))  float  v8f;

union BFrag { v16bf v; unsigned u[8]; };   // 16 bf16 = 8 dwords = 8 VGPRs
union CFrag { v8f  v; float    f[8]; };    // 8 f32 accumulator VGPRs

__device__ __forceinline__ unsigned short f2bf(float x) {
  unsigned u = __builtin_bit_cast(unsigned, x);
  u += 0x7FFFu + ((u >> 16) & 1u);         // round-to-nearest-even
  return (unsigned short)(u >> 16);
}

// ---------------------------------------------------------------------------
__global__ void lstm_cvt_bf16(const float* __restrict__ src,
                              unsigned short* __restrict__ dst, int n) {
  int stride = gridDim.x * blockDim.x;
  for (int i = blockIdx.x * blockDim.x + threadIdx.x; i < n; i += stride)
    dst[i] = f2bf(src[i]);
}

// ---------------------------------------------------------------------------
// xg[dir][row][g] = sum_k x[row][k]*Wih[dir][g][k] + bih[g] + bhh[g]
// one 16x16 tile per wave, K=256 in 8 wmma steps.
__global__ __launch_bounds__(256) void lstm_gemm_xg(
    const unsigned short* __restrict__ xb,    // [32768][256] bf16
    const unsigned short* __restrict__ wih,   // [2][512][256] bf16
    const float* __restrict__ bih_f, const float* __restrict__ bhh_f,
    const float* __restrict__ bih_b, const float* __restrict__ bhh_b,
    float* __restrict__ xg)                   // [2][32768][512] f32
{
  const int lane = threadIdx.x & 31, wave = threadIdx.x >> 5;
  const int half = lane >> 4, lm = lane & 15;
  const int gidx  = blockIdx.x * 8 + wave;        // 131072 tiles total
  const int dir   = gidx >> 16;                   // 65536 tiles per dir
  const int rem   = gidx & 65535;
  const int mtile = rem >> 5, ntile = rem & 31;
  const int n     = ntile * 16 + lm;

  const float* bi = dir ? bih_b : bih_f;
  const float* bh = dir ? bhh_b : bhh_f;
  const float  bias = bi[n] + bh[n];              // same for all 8 C elements

  CFrag acc;
  #pragma unroll
  for (int r = 0; r < 8; ++r) acc.f[r] = bias;

  const int row = mtile * 16 + lm;                // token row (b*T+t)
  const unsigned short* arow = xb  + (size_t)row * LSTM_D;
  const unsigned short* brow = wih + ((size_t)dir * LSTM_G + n) * LSTM_D;

  #pragma unroll
  for (int kk = 0; kk < 8; ++kk) {                // K = 8*32 = 256
    BFrag a, b;
    #pragma unroll
    for (int p = 0; p < 8; ++p) {
      // A 16x32 bf16: elem e=2p -> K = 2p + (p>=4?8:0) + half*8
      a.u[p] = *(const unsigned*)(arow + kk*32 + 2*p + (p >= 4 ? 8 : 0) + half*8);
      // B 32x16 bf16: elem e=2p -> K = 2p + (p>=4?8:0) + half*8; B[k][n]=Wih[n][k]
      b.u[p] = *(const unsigned*)(brow + kk*32 + 2*p + (p >= 4 ? 8 : 0) + half*8);
    }
    acc.v = __builtin_amdgcn_wmma_f32_16x16x32_bf16(
        false, a.v, false, b.v, (short)0, acc.v, false, false);
  }
  #pragma unroll
  for (int r = 0; r < 8; ++r) {
    int rr = mtile * 16 + r + half * 8;           // C: M = r + 8*(lane/16)
    xg[((size_t)dir * 32768 + rr) * LSTM_G + ntile * 16 + lm] = acc.f[r];
  }
}

// ---------------------------------------------------------------------------
// Sequential scan. grid = 8 WGs (dir = wg/4, batch slice = (wg%4)*16),
// 512 threads = 16 waves; wave w owns gate N-tiles {2w, 2w+1}.
__global__ __launch_bounds__(512) void lstm_scan(
    const float* __restrict__ xg,                 // [2][32768][512]
    const unsigned short* __restrict__ whh,       // [2][512][128] bf16
    const int* __restrict__ lengths,
    float* __restrict__ out)                      // out(B,T,256) | h_n | c_n
{
  __shared__ float          gates[16 * GSTR];
  __shared__ unsigned short hbuf [16 * HSTR];

  const int tid  = threadIdx.x;
  const int lane = tid & 31, wave = tid >> 5;
  const int half = lane >> 4, lm = lane & 15;
  const int wg   = blockIdx.x;
  const int dir  = wg >> 2;
  const int b0   = (wg & 3) << 4;

  // --- Whh B-fragments pinned in VGPRs for the entire scan (64 VGPRs) -----
  BFrag bfr[2][4];
  #pragma unroll
  for (int i = 0; i < 2; ++i) {
    const int nt = wave * 2 + i;
    const unsigned short* wrow =
        whh + ((size_t)dir * LSTM_G + nt * 16 + lm) * LSTM_HC;  // B[k][n]=Whh[n][k]
    #pragma unroll
    for (int kk = 0; kk < 4; ++kk)
      #pragma unroll
      for (int p = 0; p < 8; ++p)
        bfr[i][kk].u[p] = *(const unsigned*)(wrow + kk*32 + 2*p + (p >= 4 ? 8 : 0) + half*8);
  }

  for (int i = tid; i < 16 * HSTR; i += 512) hbuf[i] = 0;   // h0 = 0

  // --- per-thread elementwise state: rows m0+4k, column j ------------------
  const int m0 = tid >> 7;        // 0..3
  const int j  = tid & 127;
  float hreg[4], creg[4];
  int   len[4];
  #pragma unroll
  for (int k = 0; k < 4; ++k) {
    hreg[k] = 0.f; creg[k] = 0.f;
    len[k]  = lengths[b0 + m0 + 4 * k];
  }
  __syncthreads();

  for (int t = 0; t < LSTM_T; ++t) {
    const int tt = dir ? (LSTM_T - 1 - t) : t;

    // A fragments: h_prev (16 x 128 bf16) from LDS, 4 K-blocks of 32
    BFrag afr[4];
    #pragma unroll
    for (int kk = 0; kk < 4; ++kk)
      #pragma unroll
      for (int p = 0; p < 8; ++p) {
        int K = kk*32 + 2*p + (p >= 4 ? 8 : 0) + half*8;
        afr[kk].u[p] = *(const unsigned*)(&hbuf[lm * HSTR + K]);
      }

    #pragma unroll
    for (int i = 0; i < 2; ++i) {
      const int nt = wave * 2 + i;
      CFrag acc;
      #pragma unroll
      for (int r = 0; r < 8; ++r) {               // C init = xg_t (has biases)
        int b = b0 + r + half * 8;
        acc.f[r] = xg[((size_t)dir * 32768 + (size_t)b * LSTM_T + tt) * LSTM_G
                      + nt * 16 + lm];
      }
      #pragma unroll
      for (int kk = 0; kk < 4; ++kk)
        acc.v = __builtin_amdgcn_wmma_f32_16x16x32_bf16(
            false, afr[kk].v, false, bfr[i][kk].v, (short)0, acc.v, false, false);
      #pragma unroll
      for (int r = 0; r < 8; ++r)
        gates[(r + half * 8) * GSTR + nt * 16 + lm] = acc.f[r];
    }

    // prefetch next step's xg rows (global_prefetch_b8)
    if (t + 1 < LSTM_T) {
      const int ttn = dir ? (LSTM_T - 2 - t) : (t + 1);
      const int b   = b0 + lm;
      __builtin_prefetch(
          &xg[((size_t)dir * 32768 + (size_t)b * LSTM_T + ttn) * LSTM_G
              + wave * 32 + half * 16], 0, 1);
    }
    __syncthreads();

    // elementwise gate math, state update, masking, output
    #pragma unroll
    for (int k = 0; k < 4; ++k) {
      const int m = m0 + 4 * k;
      float gi = gates[m * GSTR + j];
      float gf = gates[m * GSTR + 128 + j];
      float gg = gates[m * GSTR + 256 + j];
      float go = gates[m * GSTR + 384 + j];
      float i_ = 1.f / (1.f + __expf(-gi));
      float f_ = 1.f / (1.f + __expf(-gf));
      float g_ = tanhf(gg);
      float o_ = 1.f / (1.f + __expf(-go));
      float cn = f_ * creg[k] + i_ * g_;
      float hn = o_ * tanhf(cn);
      const bool act = tt < len[k];
      creg[k] = act ? cn : creg[k];
      hreg[k] = act ? hn : hreg[k];
      // out = hs * mask  ->  h_new when active, 0 past the sequence end
      out[(((size_t)(b0 + m)) * LSTM_T + tt) * 256 + dir * LSTM_HC + j] =
          act ? hn : 0.f;
      hbuf[m * HSTR + j] = f2bf(hreg[k]);
    }
    __syncthreads();
  }

  // final h_n / c_n
  const size_t outN = (size_t)LSTM_B * LSTM_T * 256;
  #pragma unroll
  for (int k = 0; k < 4; ++k) {
    const int b = b0 + m0 + 4 * k;
    out[outN +         ((size_t)dir * LSTM_B + b) * LSTM_HC + j] = hreg[k];
    out[outN + 16384 + ((size_t)dir * LSTM_B + b) * LSTM_HC + j] = creg[k];
  }
}

// ---------------------------------------------------------------------------
extern "C" void kernel_launch(void* const* d_in, const int* in_sizes, int n_in,
                              void* d_out, int out_size, void* d_ws, size_t ws_size,
                              hipStream_t stream) {
  (void)in_sizes; (void)n_in; (void)out_size; (void)ws_size;
  const float* x      = (const float*)d_in[0];
  const int*   lens   = (const int*)  d_in[1];
  const float* Wih_f  = (const float*)d_in[2];
  const float* Whh_f  = (const float*)d_in[3];
  const float* bih_f  = (const float*)d_in[4];
  const float* bhh_f  = (const float*)d_in[5];
  const float* Wih_b  = (const float*)d_in[6];
  const float* Whh_b  = (const float*)d_in[7];
  const float* bih_b  = (const float*)d_in[8];
  const float* bhh_b  = (const float*)d_in[9];
  float* out = (float*)d_out;

  char* ws = (char*)d_ws;
  unsigned short* xb  = (unsigned short*)(ws);                       // 16 MB
  unsigned short* wih = (unsigned short*)(ws + 16777216);            // 512 KB
  unsigned short* whh = (unsigned short*)(ws + 16777216 + 524288);   // 256 KB
  float*          xg  = (float*)(ws + 16777216 + 524288 + 262144);   // 128 MB

  lstm_cvt_bf16<<<2048, 256, 0, stream>>>(x,     xb,           8388608);
  lstm_cvt_bf16<<<128,  256, 0, stream>>>(Wih_f, wih,          131072);
  lstm_cvt_bf16<<<128,  256, 0, stream>>>(Wih_b, wih + 131072, 131072);
  lstm_cvt_bf16<<<64,   256, 0, stream>>>(Whh_f, whh,          65536);
  lstm_cvt_bf16<<<64,   256, 0, stream>>>(Whh_b, whh + 65536,  65536);

  lstm_gemm_xg<<<16384, 256, 0, stream>>>(xb, wih, bih_f, bhh_f, bih_b, bhh_b, xg);
  lstm_scan<<<8, 512, 0, stream>>>(xg, whh, lens, out);
}
